// TaskModel_34617436405988
// MI455X (gfx1250) — compile-verified, matrix operand order checked by reading, not run on hardware
//
#include <hip/hip_runtime.h>

// MoE routed linear (1-of-8 experts) as a grouped bf16 WMMA GEMM for gfx1250.
//   x:[8192,2048] f32, masks:[8192] i32, W:[8,2048,2048] f32, b:[8,2048] f32
//   out:[8192,2048] f32
// Pipeline: histogram -> scan/tile-table -> scatter(perm) -> grouped GEMM
// GEMM: 128x128x32 block tiles, 8 waves x (2x4) v_wmma_f32_16x16x32_bf16,
// f32 global loads converted to bf16 while staging into double-buffered LDS.
// Padding rows are clamped to a valid in-group row (A-row m only feeds D-row m,
// and padding rows are never stored), so the inner loop is branch-free.

typedef __attribute__((ext_vector_type(16))) __bf16 v16bf;
typedef __attribute__((ext_vector_type(8)))  float  v8f;

#define NB        8192
#define D_IN      2048
#define D_OUT     2048
#define N_EXP     8
#define BM        128
#define BN        128
#define BK        32
#define KSTEPS    (D_IN / BK)          // 64
#define LDSW      40                   // BK + 8 bf16 pad -> conflict-free b128 reads
#define MAX_RT    (NB / BM + N_EXP)    // 72 (>= sum ceil(cnt_e/BM) = 71 max)

// workspace int layout
#define WS_COUNTS   0    // [8]
#define WS_OFFSETS  8    // [9]
#define WS_CURSORS  24   // [8]
#define WS_NTILES   32   // [1]
#define WS_TEXP     64   // [MAX_RT]
#define WS_TROW     160  // [MAX_RT]
#define WS_PERM     256  // [NB]

__global__ void moe_zero(int* ws) {
    if (threadIdx.x < 8) ws[WS_COUNTS + threadIdx.x] = 0;
}

__global__ void moe_hist(const int* __restrict__ masks, int* ws) {
    int i = blockIdx.x * blockDim.x + threadIdx.x;
    if (i < NB) atomicAdd(&ws[WS_COUNTS + masks[i]], 1);
}

__global__ void moe_scan(int* ws) {
    if (threadIdx.x != 0) return;
    int off = 0, T = 0;
    for (int e = 0; e < N_EXP; ++e) {
        int c = ws[WS_COUNTS + e];
        ws[WS_OFFSETS + e] = off;
        ws[WS_CURSORS + e] = off;
        int nt = (c + BM - 1) / BM;
        for (int t = 0; t < nt; ++t) {
            ws[WS_TEXP + T] = e;
            ws[WS_TROW + T] = off + t * BM;
            ++T;
        }
        off += c;
    }
    ws[WS_OFFSETS + N_EXP] = off;   // == NB
    ws[WS_NTILES] = T;
}

__global__ void moe_scatter(const int* __restrict__ masks, int* ws) {
    int i = blockIdx.x * blockDim.x + threadIdx.x;
    if (i < NB) {
        int e = masks[i];
        int pos = atomicAdd(&ws[WS_CURSORS + e], 1);
        ws[WS_PERM + pos] = i;
    }
}

__device__ __forceinline__ v16bf ldsFrag(const __bf16* tile, int row0, int lane) {
    // 16-bit A/B fragment layout: lane = idx + 16*h holds row/col `idx`,
    // K chunks [8h, 8h+8) in VGPR0-3 and [16+8h, 16+8h+8) in VGPR4-7.
    int idx = lane & 15, h = lane >> 4;
    const __bf16* r = tile + (row0 + idx) * LDSW;
    union { v16bf v; uint4 q[2]; } u;
    u.q[0] = *(const uint4*)(r + 8 * h);
    u.q[1] = *(const uint4*)(r + 16 + 8 * h);
    return u.v;
}

__global__ __launch_bounds__(256) void moe_gemm(
        const float* __restrict__ x, const float* __restrict__ W,
        const float* __restrict__ bias, const int* __restrict__ ws,
        float* __restrict__ out) {
    const int numTiles = ws[WS_NTILES];
    const int t = blockIdx.x;
    if (t >= numTiles) return;

    const int e        = ws[WS_TEXP + t];
    const int rowBase  = ws[WS_TROW + t];          // index into perm
    const int groupEnd = ws[WS_OFFSETS + e + 1];
    const int colBase  = blockIdx.y * BN;

    __shared__ __align__(16) __bf16 As[2][BM][LDSW];
    __shared__ __align__(16) __bf16 Bs[2][BM][LDSW];

    const int tid  = threadIdx.x;
    const int lr   = tid >> 1;       // 0..127: row of the tile this thread stages
    const int half = tid & 1;        // which 16-element K chunk
    const int wave = tid >> 5;       // 0..7
    const int lane = tid & 31;
    const int wm   = wave & 3;       // wave row: rows [32*wm, 32*wm+32)
    const int wn   = wave >> 2;      // wave col: cols [64*wn, 64*wn+64)

    // Gather pointer. Padding rows are clamped to the last valid row of the
    // group (always exists: rowBase < groupEnd for every scheduled tile).
    // Their results are garbage but are never stored -> branch-free k-loop.
    int rowIdx = rowBase + lr;
    if (rowIdx >= groupEnd) rowIdx = groupEnd - 1;
    const float* xrow = x + (size_t)ws[WS_PERM + rowIdx] * D_IN;
    const float* wrow = W + ((size_t)e * D_OUT + (colBase + lr)) * D_IN;

    float4 Af[4], Bf[4];

    auto fetch = [&](int ks) {
        const float4* pa = (const float4*)(xrow + ks * BK + half * 16);
        const float4* pb = (const float4*)(wrow + ks * BK + half * 16);
#pragma unroll
        for (int q = 0; q < 4; ++q) { Af[q] = pa[q]; Bf[q] = pb[q]; }
    };

    auto stage = [&](int buf) {
        union { __bf16 h[16]; uint4 q[2]; } ua, ub;
#pragma unroll
        for (int q = 0; q < 4; ++q) {
            ua.h[4*q+0] = (__bf16)Af[q].x; ua.h[4*q+1] = (__bf16)Af[q].y;
            ua.h[4*q+2] = (__bf16)Af[q].z; ua.h[4*q+3] = (__bf16)Af[q].w;
            ub.h[4*q+0] = (__bf16)Bf[q].x; ub.h[4*q+1] = (__bf16)Bf[q].y;
            ub.h[4*q+2] = (__bf16)Bf[q].z; ub.h[4*q+3] = (__bf16)Bf[q].w;
        }
        uint4* da = (uint4*)&As[buf][lr][half * 16];
        uint4* db = (uint4*)&Bs[buf][lr][half * 16];
        da[0] = ua.q[0]; da[1] = ua.q[1];
        db[0] = ub.q[0]; db[1] = ub.q[1];
    };

    v8f acc[2][4];
#pragma unroll
    for (int i = 0; i < 2; ++i)
#pragma unroll
        for (int j = 0; j < 4; ++j) acc[i][j] = (v8f)(0.f);

    fetch(0);
    stage(0);
    __syncthreads();

    for (int ks = 0; ks < KSTEPS; ++ks) {
        if (ks + 1 < KSTEPS) fetch(ks + 1);

        const int buf = ks & 1;
        v16bf afrag[2], bfrag[4];
#pragma unroll
        for (int i = 0; i < 2; ++i)
            afrag[i] = ldsFrag(&As[buf][0][0], 32 * wm + 16 * i, lane);
#pragma unroll
        for (int j = 0; j < 4; ++j)
            bfrag[j] = ldsFrag(&Bs[buf][0][0], 64 * wn + 16 * j, lane);
#pragma unroll
        for (int i = 0; i < 2; ++i)
#pragma unroll
            for (int j = 0; j < 4; ++j)
                acc[i][j] = __builtin_amdgcn_wmma_f32_16x16x32_bf16(
                    false, afrag[i], false, bfrag[j],
                    (short)0, acc[i][j], false, false);

        if (ks + 1 < KSTEPS) {
            __syncthreads();          // all waves done reading buf^1
            stage((ks + 1) & 1);
            __syncthreads();          // stores visible before next compute
        }
    }

    // Epilogue. C/D layout: lane = N + 16*(M>=8); VGPR v holds M = v (+8).
    const int n  = lane & 15;
    const int hh = lane >> 4;
    float bv[4];
#pragma unroll
    for (int j = 0; j < 4; ++j)
        bv[j] = bias[(size_t)e * D_OUT + colBase + 64 * wn + 16 * j + n];

#pragma unroll
    for (int i = 0; i < 2; ++i) {
#pragma unroll
        for (int v = 0; v < 8; ++v) {
            const int m  = v + 8 * hh;
            const int gr = rowBase + 32 * wm + 16 * i + m;
            if (gr < groupEnd) {
                const size_t orow = (size_t)ws[WS_PERM + gr];
#pragma unroll
                for (int j = 0; j < 4; ++j) {
                    const int col = colBase + 64 * wn + 16 * j + n;
                    out[orow * D_OUT + col] = acc[i][j][v] + bv[j];
                }
            }
        }
    }
}

extern "C" void kernel_launch(void* const* d_in, const int* in_sizes, int n_in,
                              void* d_out, int out_size, void* d_ws, size_t ws_size,
                              hipStream_t stream) {
    (void)in_sizes; (void)n_in; (void)out_size; (void)ws_size;
    const float* x     = (const float*)d_in[0];
    const int*   masks = (const int*)d_in[1];
    const float* W     = (const float*)d_in[2];
    const float* b     = (const float*)d_in[3];
    float* out = (float*)d_out;
    int*   ws  = (int*)d_ws;

    moe_zero<<<1, 64, 0, stream>>>(ws);
    moe_hist<<<NB / 256, 256, 0, stream>>>(masks, ws);
    moe_scan<<<1, 1, 0, stream>>>(ws);
    moe_scatter<<<NB / 256, 256, 0, stream>>>(masks, ws);
    moe_gemm<<<dim3(MAX_RT, D_OUT / BN), 256, 0, stream>>>(x, W, b, ws, out);
}
